// MultiHeadAttention_50002009260563
// MI455X (gfx1250) — compile-verified, matrix operand order checked by reading, not run on hardware
//
#include <hip/hip_runtime.h>
#include <hip/hip_bf16.h>

// ---------------------------------------------------------------------------
// Types for CDNA5 WMMA (wave32): v_wmma_f32_16x16x32_bf16
// ---------------------------------------------------------------------------
typedef __attribute__((ext_vector_type(16))) __bf16 v16bf;
typedef __attribute__((ext_vector_type(8)))  __bf16 v8bf;
typedef __attribute__((ext_vector_type(4)))  __bf16 v4bf;
typedef __attribute__((ext_vector_type(8)))  float  v8f;
typedef __attribute__((ext_vector_type(4)))  float  v4f;

__device__ __forceinline__ v16bf cat8(v8bf lo, v8bf hi) {
  union { v16bf v; v8bf h[2]; } u;
  u.h[0] = lo; u.h[1] = hi;
  return u.v;
}

__device__ __forceinline__ v8f wmma_bf16(v16bf a, v16bf b, v8f c) {
  // 8 args: (neg_a, A, neg_b, B, c_mod, C, reuse_a, reuse_b)
  return __builtin_amdgcn_wmma_f32_16x16x32_bf16(
      false, a, false, b, (short)0, c, false, false);
}

// Problem constants (B=4, T=2048, D=1024, H=16, HD=64)
#define TT   2048
#define HH   16
#define HDD  64

// LDS row strides (padded to dodge 64-bank conflicts; multiples of 8 elems
// keep 16B alignment for b128 fragment loads)
#define GSTRIDE 72   // GEMM tiles: 64 K-elems + 8 pad  (144B rows)
#define ASTRIDE 40   // attention tiles: 32 key-elems + 8 pad (80B rows)

// ---------------------------------------------------------------------------
// GEMM: C[M,N] = A[M,K] * Bw[K,N] + bias, bf16 WMMA with fp32 accumulate.
// Block tile 128x128, K-step 64 (two WMMA k-steps per LDS stage).
// 256 threads = 8 waves arranged 4(M) x 2(N); each wave owns 32x64 = 2x4
// WMMA accumulators.
// MODE 0: A is fp32 (x), epilogue scatters bf16 Q/K/V to [B*H,T,64] (Q scaled)
// MODE 1: A is bf16 (y_ws), epilogue stores fp32 to d_out
// ---------------------------------------------------------------------------
template <int MODE>
__global__ void __launch_bounds__(256)
gemm_kernel(const void* __restrict__ Aptr, const float* __restrict__ Bw,
            const float* __restrict__ bias,
            __bf16* __restrict__ q_ws, __bf16* __restrict__ k_ws,
            __bf16* __restrict__ v_ws, float* __restrict__ outp,
            int M, int K, int N) {
  __shared__ __bf16 As[128 * GSTRIDE];   // [row][k] row-major, padded
  __shared__ __bf16 Bs[128 * GSTRIDE];   // [col][k] (B transposed), padded

  const int tid   = threadIdx.x;
  const int wave  = tid >> 5;
  const int lane  = tid & 31;
  const int l16   = lane & 15;
  const int khalf = lane >> 4;
  const int wm    = wave >> 1;      // 0..3
  const int wn    = wave & 1;       // 0..1
  const int m0    = blockIdx.y * 128;
  const int n0    = blockIdx.x * 128;

  v8f acc[2][4];
#pragma unroll
  for (int mi = 0; mi < 2; ++mi)
#pragma unroll
    for (int ni = 0; ni < 4; ++ni)
#pragma unroll
      for (int r = 0; r < 8; ++r) acc[mi][ni][r] = 0.0f;

  for (int k0 = 0; k0 < K; k0 += 64) {
    __syncthreads();
    // ---- stage A tile (128 x 64) into LDS as bf16 ----
    if (MODE == 0) {
      const float* A = (const float*)Aptr;
#pragma unroll
      for (int i = 0; i < 8; ++i) {
        int qi  = tid + i * 256;          // 2048 float4 quads
        int row = qi >> 4;                // 16 quads per row
        int c4  = (qi & 15) << 2;
        const float* src = A + (size_t)(m0 + row) * K + k0 + c4;
        v4f f = *(const v4f*)src;
        if (i == 0 && k0 + 64 < K) __builtin_prefetch(src + 64, 0, 1);
        v4bf h;
        h[0] = (__bf16)f[0]; h[1] = (__bf16)f[1];
        h[2] = (__bf16)f[2]; h[3] = (__bf16)f[3];
        *(v4bf*)&As[row * GSTRIDE + c4] = h;
      }
    } else {
      const __bf16* A = (const __bf16*)Aptr;
#pragma unroll
      for (int i = 0; i < 4; ++i) {
        int oi  = tid + i * 256;          // 1024 8-element groups
        int row = oi >> 3;
        int c8  = (oi & 7) << 3;
        const __bf16* src = A + (size_t)(m0 + row) * K + k0 + c8;
        if (i == 0 && k0 + 64 < K) __builtin_prefetch(src + 64, 0, 1);
        *(v8bf*)&As[row * GSTRIDE + c8] = *(const v8bf*)src;
      }
    }
    // ---- stage B tile (64 x 128) transposed into LDS ----
#pragma unroll
    for (int i = 0; i < 8; ++i) {
      int qi = tid + i * 256;             // 2048 float4 quads
      int kr = qi >> 5;                   // 32 quads per k-row
      int c4 = (qi & 31) << 2;
      const float* src = Bw + (size_t)(k0 + kr) * N + n0 + c4;
      v4f f = *(const v4f*)src;
      if (i == 0 && k0 + 64 < K) __builtin_prefetch(src + (size_t)64 * N, 0, 1);
      Bs[(c4 + 0) * GSTRIDE + kr] = (__bf16)f[0];
      Bs[(c4 + 1) * GSTRIDE + kr] = (__bf16)f[1];
      Bs[(c4 + 2) * GSTRIDE + kr] = (__bf16)f[2];
      Bs[(c4 + 3) * GSTRIDE + kr] = (__bf16)f[3];
    }
    __syncthreads();

    // ---- fragments + WMMA: two k-steps per stage ----
#pragma unroll
    for (int kk = 0; kk < 64; kk += 32) {
      v16bf af[2], bfv[4];
#pragma unroll
      for (int mi = 0; mi < 2; ++mi) {
        int row = wm * 32 + mi * 16 + l16;
        const __bf16* p = &As[row * GSTRIDE + kk + khalf * 8];
        af[mi] = cat8(*(const v8bf*)p, *(const v8bf*)(p + 16));
      }
#pragma unroll
      for (int ni = 0; ni < 4; ++ni) {
        int col = wn * 64 + ni * 16 + l16;
        const __bf16* p = &Bs[col * GSTRIDE + kk + khalf * 8];
        bfv[ni] = cat8(*(const v8bf*)p, *(const v8bf*)(p + 16));
      }
#pragma unroll
      for (int mi = 0; mi < 2; ++mi)
#pragma unroll
        for (int ni = 0; ni < 4; ++ni)
          acc[mi][ni] = wmma_bf16(af[mi], bfv[ni], acc[mi][ni]);
    }
  }

  // ---- epilogue (C layout: elem r, lane L -> M = r + 8*(L/16), N = L%16) ----
#pragma unroll
  for (int mi = 0; mi < 2; ++mi) {
#pragma unroll
    for (int ni = 0; ni < 4; ++ni) {
      int col = n0 + wn * 64 + ni * 16 + l16;
      float bv = bias[col];
#pragma unroll
      for (int r = 0; r < 8; ++r) {
        int row = m0 + wm * 32 + mi * 16 + r + 8 * khalf;
        float val = acc[mi][ni][r] + bv;
        if (MODE == 0) {
          int which = col >> 10;              // 0=q,1=k,2=v
          int rem   = col & 1023;
          int h     = rem >> 6;
          int hd    = rem & 63;
          int b     = row >> 11;              // T = 2048
          int t     = row & 2047;
          size_t dst = (((size_t)(b * HH + h)) * TT + t) * HDD + hd;
          __bf16* base = (which == 0) ? q_ws : (which == 1) ? k_ws : v_ws;
          base[dst] = (__bf16)((which == 0) ? val * 0.125f : val);
        } else {
          outp[(size_t)row * N + col] = val;
        }
      }
    }
  }
}

// ---------------------------------------------------------------------------
// Flash attention: grid (B*H, T/128), 256 threads = 8 waves; wave w owns
// q rows [qblk*128 + w*16, +16). Iterates 32-key tiles up to causal limit.
// Q/K/V in bf16 [B*H, T, 64]; Q pre-scaled by 1/sqrt(HD).
// ---------------------------------------------------------------------------
__global__ void __launch_bounds__(256)
attn_kernel(const __bf16* __restrict__ Q, const __bf16* __restrict__ K,
            const __bf16* __restrict__ V, __bf16* __restrict__ Y) {
  __shared__ __bf16 Vt[64 * ASTRIDE];        // V transposed: [hd][key], padded
  __shared__ __bf16 Pt[8 * 16 * ASTRIDE];    // per-wave P tile, [qrow][key]

  const int bh    = blockIdx.x;         // 0..63
  const int qblk  = blockIdx.y;         // 0..15
  const int tid   = threadIdx.x;
  const int wave  = tid >> 5;
  const int lane  = tid & 31;
  const int l16   = lane & 15;
  const int khalf = lane >> 4;
  const int qbase = qblk * 128 + wave * 16;

  const size_t base = (size_t)bh * TT * HDD;
  const __bf16* Qb = Q + base;
  const __bf16* Kb = K + base;
  const __bf16* Vb = V + base;

  // Q fragments (A layout, 16x32): two HD slices
  const __bf16* qp = Qb + (size_t)(qbase + l16) * HDD;
  v16bf qf0 = cat8(*(const v8bf*)(qp + khalf * 8),
                   *(const v8bf*)(qp + 16 + khalf * 8));
  v16bf qf1 = cat8(*(const v8bf*)(qp + 32 + khalf * 8),
                   *(const v8bf*)(qp + 48 + khalf * 8));

  v8f o[4];
  float mrow[8], lrow[8];
#pragma unroll
  for (int n = 0; n < 4; ++n)
#pragma unroll
    for (int r = 0; r < 8; ++r) o[n][r] = 0.0f;
#pragma unroll
  for (int r = 0; r < 8; ++r) { mrow[r] = -1e30f; lrow[r] = 0.0f; }

  const int nkt = (qblk + 1) * 4;       // 32-key tiles, uniform per block
  for (int kt = 0; kt < nkt; ++kt) {
    const int kbase = kt * 32;
    // prefetch next tile's K/V rows into cache while this tile computes
    if (kt + 1 < nkt) {
      __builtin_prefetch(Kb + (size_t)(kbase + 32 + l16) * HDD, 0, 1);
      __builtin_prefetch(Vb + (size_t)(kbase + 32 + l16) * HDD, 0, 1);
    }
    __syncthreads();
    // ---- stage V tile transposed: Vt[hd][key] ----
    {
      int key = tid >> 3;
      int hg  = (tid & 7) << 3;
      v8bf vv = *(const v8bf*)(Vb + (size_t)(kbase + key) * HDD + hg);
#pragma unroll
      for (int j = 0; j < 8; ++j) Vt[(hg + j) * ASTRIDE + key] = vv[j];
    }
    __syncthreads();

    // ---- S = Q * K^T for 32 keys (two 16-key column tiles) ----
    const __bf16* kp0 = Kb + (size_t)(kbase + l16) * HDD;
    v16bf k0a = cat8(*(const v8bf*)(kp0 + khalf * 8),
                     *(const v8bf*)(kp0 + 16 + khalf * 8));
    v16bf k0b = cat8(*(const v8bf*)(kp0 + 32 + khalf * 8),
                     *(const v8bf*)(kp0 + 48 + khalf * 8));
    v8f s0;
#pragma unroll
    for (int r = 0; r < 8; ++r) s0[r] = 0.0f;
    s0 = wmma_bf16(qf0, k0a, s0);
    s0 = wmma_bf16(qf1, k0b, s0);

    const __bf16* kp1 = Kb + (size_t)(kbase + 16 + l16) * HDD;
    v16bf k1a = cat8(*(const v8bf*)(kp1 + khalf * 8),
                     *(const v8bf*)(kp1 + 16 + khalf * 8));
    v16bf k1b = cat8(*(const v8bf*)(kp1 + 32 + khalf * 8),
                     *(const v8bf*)(kp1 + 48 + khalf * 8));
    v8f s1;
#pragma unroll
    for (int r = 0; r < 8; ++r) s1[r] = 0.0f;
    s1 = wmma_bf16(qf0, k1a, s1);
    s1 = wmma_bf16(qf1, k1b, s1);

    // ---- causal mask + online softmax (rows across 16-lane halves) ----
    float al[8];
#pragma unroll
    for (int r = 0; r < 8; ++r) {
      int qr = qbase + r + 8 * khalf;
      float v0 = (kbase + l16 <= qr)      ? (float)s0[r] : -1e30f;
      float v1 = (kbase + 16 + l16 <= qr) ? (float)s1[r] : -1e30f;
      float mx = fmaxf(v0, v1);
#pragma unroll
      for (int d = 1; d < 16; d <<= 1) mx = fmaxf(mx, __shfl_xor(mx, d, 32));
      float mn = fmaxf(mrow[r], mx);
      float a  = __expf(mrow[r] - mn);
      mrow[r]  = mn;
      float p0 = __expf(v0 - mn);
      float p1 = __expf(v1 - mn);
      float sm = p0 + p1;
#pragma unroll
      for (int d = 1; d < 16; d <<= 1) sm += __shfl_xor(sm, d, 32);
      lrow[r] = lrow[r] * a + sm;
      al[r]   = a;
      __bf16* pr = &Pt[(wave * 16 + r + 8 * khalf) * ASTRIDE];
      pr[l16]      = (__bf16)p0;
      pr[l16 + 16] = (__bf16)p1;
    }
#pragma unroll
    for (int n = 0; n < 4; ++n)
#pragma unroll
      for (int r = 0; r < 8; ++r) o[n][r] *= al[r];

    __syncthreads();   // P tile visible; uniform trip count per block

    // ---- O += P * V (P as A 16x32, V^T rows give B fragments) ----
    const __bf16* pp = &Pt[(wave * 16 + l16) * ASTRIDE + khalf * 8];
    v16bf pf = cat8(*(const v8bf*)pp, *(const v8bf*)(pp + 16));
#pragma unroll
    for (int n = 0; n < 4; ++n) {
      const __bf16* vp = &Vt[(n * 16 + l16) * ASTRIDE + khalf * 8];
      v16bf vf = cat8(*(const v8bf*)vp, *(const v8bf*)(vp + 16));
      o[n] = wmma_bf16(pf, vf, o[n]);
    }
  }

  // ---- epilogue: normalize, store y as bf16 [B, T, H*HD] ----
  const int b = bh >> 4;
  const int h = bh & 15;
#pragma unroll
  for (int n = 0; n < 4; ++n) {
#pragma unroll
    for (int r = 0; r < 8; ++r) {
      int qr = qbase + r + 8 * khalf;
      float val = o[n][r] / lrow[r];
      size_t dst = ((size_t)(b * TT + qr)) * (HH * HDD) + h * HDD + n * 16 + l16;
      Y[dst] = (__bf16)val;
    }
  }
}

// ---------------------------------------------------------------------------
// Launch: QKV GEMM -> flash attention -> output GEMM (all on `stream`)
// ---------------------------------------------------------------------------
extern "C" void kernel_launch(void* const* d_in, const int* in_sizes, int n_in,
                              void* d_out, int out_size, void* d_ws, size_t ws_size,
                              hipStream_t stream) {
  const float* x     = (const float*)d_in[0];
  // d_in[1] is the causal mask; handled analytically in attn_kernel.
  const float* w_qkv = (const float*)d_in[2];
  const float* b_qkv = (const float*)d_in[3];
  const float* w_out = (const float*)d_in[4];
  const float* b_out = (const float*)d_in[5];
  float* out = (float*)d_out;

  const int Bn = 4, Dn = 1024;
  const int Mrows = Bn * TT;                       // 8192
  const size_t qkvE = (size_t)Bn * HH * TT * HDD;  // 8,388,608 elems

  __bf16* ws  = (__bf16*)d_ws;
  __bf16* qws = ws;
  __bf16* kws = ws + qkvE;
  __bf16* vws = ws + 2 * qkvE;
  __bf16* yws = ws + 3 * qkvE;                     // [8192, 1024] bf16

  // 1) qkv = x @ w_qkv + b_qkv  -> bf16 Q/K/V in [B*H, T, 64] (Q * 1/8)
  gemm_kernel<0><<<dim3((3 * Dn) / 128, Mrows / 128), 256, 0, stream>>>(
      x, w_qkv, b_qkv, qws, kws, vws, nullptr, Mrows, Dn, 3 * Dn);

  // 2) flash attention -> y_ws bf16 [B*T, D]
  attn_kernel<<<dim3(Bn * HH, TT / 128), 256, 0, stream>>>(qws, kws, vws, yws);

  // 3) out = y @ w_out + b_out  (fp32)
  gemm_kernel<1><<<dim3(Dn / 128, Mrows / 128), 256, 0, stream>>>(
      yws, w_out, b_out, nullptr, nullptr, nullptr, out, Mrows, Dn, Dn);
}
